// GAT_9947144257800
// MI455X (gfx1250) — matrix operand order repaired and checked by reading, not compiled, not run on hardware
//
#include <hip/hip_runtime.h>
#include <hip/hip_bf16.h>

// ---- problem constants (from reference) ----
#define BATCH   16
#define NNODE   256
#define INDIM   768
#define MEMD    300
#define HLD     320      // ld for h/x buffers (multiple of 32)
#define HIDD    64
#define NEGINF  (-1e30f)
#define LSLOPE  0.01f

#define TNP     20       // N tiles for the 300-wide GEMMs, padded to mult of 4
#define NTW     4        // N tiles per wave

typedef __attribute__((ext_vector_type(16))) __bf16 v16bf;
typedef __attribute__((ext_vector_type(8)))  float  v8f;

__device__ __forceinline__ unsigned short f2bfbits(float x) {
  // round-to-nearest-even f32 -> bf16 (bit pattern)
  unsigned u = __float_as_uint(x);
  return (unsigned short)((u + 0x7FFFu + ((u >> 16) & 1u)) >> 16);
}

// ---------------------------------------------------------------------------
// Pack A (f32, row-major MxK) into WMMA A-fragment order, bf16:
// dst[tm][kc][lane][e], 16 contiguous bf16 per lane (32B vector store).
// Zero-pads rows >= Msrc and k >= Ksrc.
// ---------------------------------------------------------------------------
__global__ void pack_a_frag(const float* __restrict__ src,
                            unsigned short* __restrict__ dst,
                            int tilesM, int KC, int Msrc, int Ksrc, int lda) {
  const int t = blockIdx.x * blockDim.x + threadIdx.x;
  const int total = tilesM * KC * 32;
  if (t >= total) return;
  const int lane = t & 31;
  const int kc = (t >> 5) % KC;
  const int tm = t / (KC * 32);
  const int half = lane >> 4, mr = lane & 15;
  const int row = tm * 16 + mr;

  unsigned short out[16];
#pragma unroll
  for (int e = 0; e < 16; ++e) {
    const int r = e >> 1, p = e & 1;
    const int kk = (r < 4 ? (2 * r + p) : (16 + 2 * (r - 4) + p)) + half * 8;
    const int k = kc * 32 + kk;
    const float v =
        (row < Msrc && k < Ksrc) ? src[(size_t)row * lda + k] : 0.0f;
    out[e] = f2bfbits(v);
  }
  unsigned short* d = dst + (size_t)t * 16;
  *(uint4*)(d + 0) = *(const uint4*)(out + 0);
  *(uint4*)(d + 8) = *(const uint4*)(out + 8);
}

// ---------------------------------------------------------------------------
// Pack B (f32, row-major KxN, optional batch) into WMMA B-fragment order:
// dst[bz][tn][kc][lane][e].  lane = column within tile; halves split K.
// ---------------------------------------------------------------------------
__global__ void pack_b_frag(const float* __restrict__ src,
                            unsigned short* __restrict__ dst,
                            int tilesN, int KC, int Ksrc, int Nsrc, int ldb,
                            long long srcBatchStride) {
  const int t = blockIdx.x * blockDim.x + threadIdx.x;
  const int total = tilesN * KC * 32;
  if (t >= total) return;
  const int bz = blockIdx.y;
  const int lane = t & 31;
  const int kc = (t >> 5) % KC;
  const int tn = t / (KC * 32);
  const int half = lane >> 4, mr = lane & 15;
  const int col = tn * 16 + mr;
  const float* s = src + (long long)bz * srcBatchStride;

  unsigned short out[16];
#pragma unroll
  for (int e = 0; e < 16; ++e) {
    const int k = kc * 32 + (e & 7) + ((e >> 3) << 4) + half * 8;
    const float v = (col < Nsrc && k < Ksrc) ? s[(size_t)k * ldb + col] : 0.0f;
    out[e] = f2bfbits(v);
  }
  unsigned short* d = dst + ((size_t)bz * total + t) * 16;
  *(uint4*)(d + 0) = *(const uint4*)(out + 0);
  *(uint4*)(d + 8) = *(const uint4*)(out + 8);
}

// ---------------------------------------------------------------------------
// GEMM on pre-packed fragments. One wave32 computes a 16 x (16*NT) C strip:
// per k-step, 1 A-fragment load is reused by NT WMMAs (NT B loads).
// Fully unrolled (KC template); no load guards; stores column-guarded.
// grid = (tilesN/NT, tilesMper, batch), block = 32.
// ---------------------------------------------------------------------------
template <int KC, int NT>
__global__ void wmma_gemm_frag(const unsigned short* __restrict__ Af,
                               const unsigned short* __restrict__ Bf,
                               const float* __restrict__ bias,
                               float* __restrict__ C,
                               int tilesMper, int Ncol, int ldc,
                               long long bBatchFrags, long long cBatchOff) {
  const int lane = threadIdx.x;      // 0..31
  const int half = lane >> 4;
  const int mr = lane & 15;
  const int tn0 = blockIdx.x * NT;
  const int tmL = blockIdx.y;
  const long long bz = blockIdx.z;
  const int tm = (int)bz * tilesMper + tmL;

  const v16bf* Ap = (const v16bf*)Af + ((size_t)tm * KC) * 32 + lane;
  const v16bf* Bp =
      (const v16bf*)Bf + bz * bBatchFrags + ((size_t)tn0 * KC) * 32 + lane;

  v8f acc[NT];
#pragma unroll
  for (int u = 0; u < NT; ++u) acc[u] = {};

#pragma unroll
  for (int kc = 0; kc < KC; ++kc) {
    const v16bf a = Ap[(size_t)kc * 32];
#pragma unroll
    for (int u = 0; u < NT; ++u) {
      const v16bf b = Bp[(size_t)(u * KC + kc) * 32];
      acc[u] = __builtin_amdgcn_wmma_f32_16x16x32_bf16(
          false, a, false, b, (short)0, acc[u], false, false);
    }
  }

  float* Cp = C + bz * cBatchOff;
  const int row0 = tmL * 16;
#pragma unroll
  for (int u = 0; u < NT; ++u) {
    const int col = (tn0 + u) * 16 + mr;
    if (col < Ncol) {
      const float bv = bias ? bias[col] : 0.0f;
#pragma unroll
      for (int r = 0; r < 8; ++r) {
        const int row = row0 + r + half * 8;
        Cp[(size_t)row * ldc + col] = acc[u][r] + bv;
      }
    }
  }
}

// ---------------------------------------------------------------------------
// e[b,i,j] = leakyrelu( sum_h relu(si[b,i,h]+sj[b,j,h]) * a2[h] + ab2 )
// logits[b,i,j] = e*adj + (1-adj)*NEG.  grid=(N, B), block=N (thread=j).
// ---------------------------------------------------------------------------
__global__ void edge_scores(const float* __restrict__ si,
                            const float* __restrict__ sj,
                            const float* __restrict__ a2,
                            const float* __restrict__ ab2,
                            const float* __restrict__ adj,
                            float* __restrict__ logits) {
  const int i = blockIdx.x;
  const int b = blockIdx.y;
  const int j = threadIdx.x;

  __shared__ float ssi[HIDD];
  __shared__ float sa2[HIDD];
  if (j < HIDD) {
    ssi[j] = si[((long long)(b * NNODE + i)) * HIDD + j];
    sa2[j] = a2[j];
  }
  __syncthreads();

  const float4* sjr = (const float4*)(sj + ((long long)(b * NNODE + j)) * HIDD);
  float acc = 0.0f;
#pragma unroll
  for (int h4 = 0; h4 < HIDD / 4; ++h4) {
    const float4 v = sjr[h4];
    const int h = h4 * 4;
    acc += fmaxf(ssi[h + 0] + v.x, 0.0f) * sa2[h + 0];
    acc += fmaxf(ssi[h + 1] + v.y, 0.0f) * sa2[h + 1];
    acc += fmaxf(ssi[h + 2] + v.z, 0.0f) * sa2[h + 2];
    acc += fmaxf(ssi[h + 3] + v.w, 0.0f) * sa2[h + 3];
  }
  float e = acc + ab2[0];
  e = e > 0.0f ? e : LSLOPE * e;
  const long long idx = ((long long)(b * NNODE + i)) * NNODE + j;
  const float a = adj[idx];
  logits[idx] = e * a + (1.0f - a) * NEGINF;
}

// Global (per-batch, flattened NxN) softmax stats.  grid=B, block=256.
__global__ void softmax_stats(const float* __restrict__ logits,
                              float* __restrict__ red) {
  const int b = blockIdx.x;
  const int tid = threadIdx.x;
  const float* p = logits + (long long)b * NNODE * NNODE;
  __shared__ float s[256];

  float m = -__builtin_inff();
  for (int i = tid; i < NNODE * NNODE; i += 256) m = fmaxf(m, p[i]);
  s[tid] = m;
  __syncthreads();
  for (int off = 128; off > 0; off >>= 1) {
    if (tid < off) s[tid] = fmaxf(s[tid], s[tid + off]);
    __syncthreads();
  }
  const float mx = s[0];
  __syncthreads();

  float sum = 0.0f;
  for (int i = tid; i < NNODE * NNODE; i += 256) sum += __expf(p[i] - mx);
  s[tid] = sum;
  __syncthreads();
  for (int off = 128; off > 0; off >>= 1) {
    if (tid < off) s[tid] += s[tid + off];
    __syncthreads();
  }
  if (tid == 0) {
    red[b * 2 + 0] = mx;
    red[b * 2 + 1] = s[0];
  }
}

// att = exp(logit - mx) / sum.  grid=(N, B), block=N.
__global__ void softmax_norm(float* __restrict__ att,
                             const float* __restrict__ red) {
  const int b = blockIdx.y;
  const long long i = (long long)blockIdx.x * NNODE + threadIdx.x;
  const float mx = red[b * 2 + 0];
  const float inv = 1.0f / red[b * 2 + 1];
  float* p = att + (long long)b * NNODE * NNODE;
  p[i] = __expf(p[i] - mx) * inv;
}

// ---------------------------------------------------------------------------
extern "C" void kernel_launch(void* const* d_in, const int* in_sizes, int n_in,
                              void* d_out, int out_size, void* d_ws,
                              size_t ws_size, hipStream_t stream) {
  (void)in_sizes; (void)n_in; (void)out_size; (void)ws_size;

  const float* adj     = (const float*)d_in[0];
  const float* feature = (const float*)d_in[1];
  const float* W0      = (const float*)d_in[2];
  const float* b0      = (const float*)d_in[3];
  const float* W1      = (const float*)d_in[4];
  const float* b1      = (const float*)d_in[5];
  const float* A1      = (const float*)d_in[6];
  const float* ab1     = (const float*)d_in[7];
  const float* A2      = (const float*)d_in[8];
  const float* ab2     = (const float*)d_in[9];
  float* out = (float*)d_out;

  const long long MROWS = (long long)BATCH * NNODE;  // 4096
  const int TM = (int)(MROWS / 16);                  // 256 row tiles
  const int TN_HID = HIDD / 16;                      // 4
  const int KC0 = INDIM / 32;                        // 24
  const int KC1 = HLD / 32;                          // 10 (K=300 padded to 320)
  const int KC2 = NNODE / 32;                        // 8

  // ---- workspace layout ----
  float* ws   = (float*)d_ws;
  float* bufH = ws;                                    // 4096 x 320
  float* bufX = bufH + MROWS * HLD;                    // 4096 x 320
  float* si   = bufX + MROWS * HLD;                    // 4096 x 64
  float* sj   = si + MROWS * HIDD;                     // 4096 x 64
  float* att  = sj + MROWS * HIDD;                     // 16 x 256 x 256
  float* red  = att + (long long)BATCH * NNODE * NNODE;
  unsigned short* us = (unsigned short*)(red + 32);    // 32B-aligned
  unsigned short* featf = us;                              // 256*24*512
  unsigned short* xf    = featf + (size_t)TM * KC0 * 512;  // 256*10*512
  unsigned short* hAf   = xf    + (size_t)TM * KC1 * 512;  // 256*10*512
  unsigned short* attAf = hAf   + (size_t)TM * KC1 * 512;  // 256*8*512
  unsigned short* W0f   = attAf + (size_t)TM * KC2 * 512;  // 20*24*512
  unsigned short* W1f   = W0f   + (size_t)TNP * KC0 * 512; // 20*10*512
  unsigned short* A1af  = W1f   + (size_t)TNP * KC1 * 512; // 4*10*512
  unsigned short* A1bf  = A1af  + (size_t)TN_HID * KC1 * 512;
  unsigned short* hBf   = A1bf  + (size_t)TN_HID * KC1 * 512; // 16*20*8*512

  // ---- pack weights & feature (once per call) ----
  {
    int t = TNP * KC0 * 32;
    pack_b_frag<<<dim3((t + 255) / 256, 1), 256, 0, stream>>>(
        W0, W0f, TNP, KC0, INDIM, MEMD, MEMD, 0);
    t = TNP * KC1 * 32;
    pack_b_frag<<<dim3((t + 255) / 256, 1), 256, 0, stream>>>(
        W1, W1f, TNP, KC1, MEMD, MEMD, MEMD, 0);
    t = TN_HID * KC1 * 32;
    pack_b_frag<<<dim3((t + 255) / 256, 1), 256, 0, stream>>>(
        A1, A1af, TN_HID, KC1, MEMD, HIDD, HIDD, 0);
    pack_b_frag<<<dim3((t + 255) / 256, 1), 256, 0, stream>>>(
        A1 + (long long)MEMD * HIDD, A1bf, TN_HID, KC1, MEMD, HIDD, HIDD, 0);
    t = TM * KC0 * 32;
    pack_a_frag<<<(t + 255) / 256, 256, 0, stream>>>(
        feature, featf, TM, KC0, (int)MROWS, INDIM, INDIM);
  }

  for (int l = 0; l < 2; ++l) {
    const unsigned short* Xf = (l == 0) ? featf : xf;
    const float* bb = (l == 0) ? b0 : b1;

    // h = X @ W + b  (4096 x 300), 4 N-tiles per wave
    if (l == 0)
      wmma_gemm_frag<24, NTW><<<dim3(TNP / NTW, TM, 1), 32, 0, stream>>>(
          Xf, W0f, bb, bufH, TM, MEMD, HLD, 0, 0);
    else
      wmma_gemm_frag<10, NTW><<<dim3(TNP / NTW, TM, 1), 32, 0, stream>>>(
          Xf, W1f, bb, bufH, TM, MEMD, HLD, 0, 0);

    // pack h rows as A-fragments for si/sj
    {
      int t = TM * KC1 * 32;
      pack_a_frag<<<(t + 255) / 256, 256, 0, stream>>>(
          bufH, hAf, TM, KC1, (int)MROWS, MEMD, HLD);
    }
    // si = h @ Wa ; sj = h @ Wb + ab1  (4 N-tiles = whole 64-wide strip)
    wmma_gemm_frag<10, NTW><<<dim3(TN_HID / NTW, TM, 1), 32, 0, stream>>>(
        hAf, A1af, nullptr, si, TM, HIDD, HIDD, 0, 0);
    wmma_gemm_frag<10, NTW><<<dim3(TN_HID / NTW, TM, 1), 32, 0, stream>>>(
        hAf, A1bf, ab1, sj, TM, HIDD, HIDD, 0, 0);

    // attention logits + global per-batch softmax
    edge_scores<<<dim3(NNODE, BATCH), NNODE, 0, stream>>>(si, sj, A2, ab2,
                                                          adj, att);
    softmax_stats<<<BATCH, 256, 0, stream>>>(att, red);
    softmax_norm<<<dim3(NNODE, BATCH), NNODE, 0, stream>>>(att, red);

    // pack att (A) and h (batched B) fragments for x = att @ h
    {
      int t = TM * KC2 * 32;
      pack_a_frag<<<(t + 255) / 256, 256, 0, stream>>>(
          att, attAf, TM, KC2, (int)MROWS, NNODE, NNODE);
      t = TNP * KC2 * 32;
      pack_b_frag<<<dim3((t + 255) / 256, BATCH), 256, 0, stream>>>(
          bufH, hBf, TNP, KC2, NNODE, MEMD, HLD, (long long)NNODE * HLD);
    }
    float* C = (l == 0) ? bufX : out;
    const int ldc = (l == 0) ? HLD : MEMD;
    wmma_gemm_frag<8, NTW><<<dim3(TNP / NTW, NNODE / 16, BATCH), 32, 0,
                             stream>>>(
        attAf, hBf, nullptr, C, NNODE / 16, MEMD, ldc,
        (long long)TNP * KC2 * 32, (long long)NNODE * ldc);

    // pack x as A-fragments for the next layer's dense GEMM
    if (l == 0) {
      int t = TM * KC1 * 32;
      pack_a_frag<<<(t + 255) / 256, 256, 0, stream>>>(
          bufX, xf, TM, KC1, (int)MROWS, MEMD, HLD);
    }
  }
}
